// SMPL_3461743640797
// MI455X (gfx1250) — matrix operand order repaired and checked by reading, not compiled, not run on hardware
//
#include <hip/hip_runtime.h>

typedef __attribute__((ext_vector_type(16))) __bf16 v16bf;
typedef __attribute__((ext_vector_type(8)))  float  v8f;

#define NV_     6890
#define NB_     1024
#define NJ_     24
#define M1_     (NV_ * 3)        /* 20670 GEMM1 rows */
#define VTILES_ 431              /* ceil(6890/16) */
#define MTILES_ (VTILES_ * 3)    /* 1293: rows padded to 20688 */
#define KT1_    7                /* 7 K-tiles of 32 -> K = 224 */
#define K1_     224
#define BTILES_ 64               /* 1024/16 */

__constant__ int c_parent[NJ_] = {-1,0,0,0,1,2,3,4,5,6,7,8,9,9,9,12,13,14,16,17,18,19,20,21};

__device__ __forceinline__ __bf16 f2bf(float f) {
    unsigned u = __builtin_bit_cast(unsigned, f);
    u += 0x7fffu + ((u >> 16) & 1u);                 // round-to-nearest-even
    unsigned short h = (unsigned short)(u >> 16);
    return __builtin_bit_cast(__bf16, h);
}

// 16-bit A-matrix 16x32 fragment: lanes 0-15 hold M=lane, K in {0..7,16..23};
// lanes 16-31 hold M=lane-16, K in {8..15,24..31}.  e = dword*2+half (0..15).
__device__ __forceinline__ int frag_kA(int e, int half) {
    return e + ((e >= 8) ? 8 : 0) + (half ? 8 : 0);
}
// 16-bit B-matrix 32x16 fragment: lanes 0-15 = column n, K in {0..7,16..23};
// lanes 16-31 = column n, K in {8..15,24..31}.
__device__ __forceinline__ int frag_kB(int e, int half) {
    return e + ((e >= 8) ? 8 : 0) + (half ? 8 : 0);
}

__device__ __forceinline__ void rodrigues(const float* r, float* R) {
    float x = r[0], y = r[1], z = r[2];
    float th = sqrtf(x * x + y * y + z * z + 1e-8f);
    float inv = 1.0f / th;
    float kx = x * inv, ky = y * inv, kz = z * inv;
    float st = sinf(th), ct = cosf(th), oc = 1.0f - ct;
    R[0] = ct + kx * kx * oc;  R[1] = kx * ky * oc - kz * st; R[2] = kx * kz * oc + ky * st;
    R[3] = ky * kx * oc + kz * st; R[4] = ct + ky * ky * oc;  R[5] = ky * kz * oc - kx * st;
    R[6] = kz * kx * oc - ky * st; R[7] = kz * ky * oc + kx * st; R[8] = ct + kz * kz * oc;
}

// ---- 1) JT = Jreg @ v_template (24x3); JS = Jreg @ shapedirs (24x3x10) ----
__global__ void k_prep_J(const float* __restrict__ Jreg, const float* __restrict__ vtemp,
                         const float* __restrict__ shapedirs, float* __restrict__ JTJS) {
    int t = blockIdx.x * blockDim.x + threadIdx.x;
    if (t >= NJ_ * 33) return;
    int j = t / 33, c = t % 33;
    const float* row = Jreg + (size_t)j * NV_;
    float s = 0.0f;
    if (c < 3) {
        for (int v = 0; v < NV_; ++v) s += row[v] * vtemp[v * 3 + c];
        JTJS[j * 3 + c] = s;
    } else {
        int d = (c - 3) / 10, k = (c - 3) % 10;
        for (int v = 0; v < NV_; ++v) s += row[v] * shapedirs[(size_t)(v * 3 + d) * 10 + k];
        JTJS[72 + (j * 3 + d) * 10 + k] = s;
    }
}

// ---- 2) pack A = [shapedirs | posedirs] as bf16 WMMA A fragments ----
__global__ void k_pack_A(const float* __restrict__ shapedirs, const float* __restrict__ posedirs,
                         __bf16* __restrict__ Apack) {
    int t = blockIdx.x * blockDim.x + threadIdx.x;
    if (t >= MTILES_ * KT1_ * 512) return;
    int e = t & 15, l = (t >> 4) & 31, tile = t >> 9;
    int kt = tile % KT1_, mt = tile / KT1_;
    int half = l >> 4, m = l & 15;
    int gm = mt * 16 + m;
    int gk = kt * 32 + frag_kA(e, half);
    float v = 0.0f;
    if (gm < M1_) {
        if (gk < 10)        v = shapedirs[(size_t)gm * 10 + gk];
        else if (gk < 217)  v = posedirs[(size_t)gm * 207 + (gk - 10)];
    }
    Apack[t] = f2bf(v);
}

// ---- 3) per-(batch,joint): rodrigues, kinematic chain, G affines, x vector ----
__global__ void k_batch(const float* __restrict__ pose, const float* __restrict__ beta,
                        const float* __restrict__ JTJS, float* __restrict__ G,
                        float* __restrict__ xvec) {
    int t = blockIdx.x * blockDim.x + threadIdx.x;
    if (t >= NB_ * NJ_) return;
    int b = t / NJ_, j = t % NJ_;
    const float* bet = beta + b * 10;
    const float* JT = JTJS;
    const float* JS = JTJS + 72;

    auto jloc = [&](int i, float* o) {
        for (int d = 0; d < 3; ++d) {
            float s = JT[i * 3 + d];
            for (int k = 0; k < 10; ++k) s += JS[(i * 3 + d) * 10 + k] * bet[k];
            o[d] = s;
        }
    };

    int chain[10], depth = 0;
    for (int c = j; c >= 0; c = c_parent[c]) chain[depth++] = c;

    float AR[9], AT[3], Rj[9];
    for (int s = depth - 1; s >= 0; --s) {
        int i = chain[s];
        float R[9];
        rodrigues(pose + (size_t)b * 72 + i * 3, R);
        if (s == 0) for (int q = 0; q < 9; ++q) Rj[q] = R[q];
        float Ji[3], ti[3];
        jloc(i, Ji);
        if (i == 0) { ti[0] = Ji[0]; ti[1] = Ji[1]; ti[2] = Ji[2]; }
        else {
            float Jp[3]; jloc(c_parent[i], Jp);
            ti[0] = Ji[0] - Jp[0]; ti[1] = Ji[1] - Jp[1]; ti[2] = Ji[2] - Jp[2];
        }
        if (s == depth - 1) {
            for (int q = 0; q < 9; ++q) AR[q] = R[q];
            AT[0] = ti[0]; AT[1] = ti[1]; AT[2] = ti[2];
        } else {
            float NR[9], NT[3];
            for (int r = 0; r < 3; ++r) {
                for (int c = 0; c < 3; ++c)
                    NR[r * 3 + c] = AR[r * 3 + 0] * R[0 * 3 + c] + AR[r * 3 + 1] * R[1 * 3 + c] + AR[r * 3 + 2] * R[2 * 3 + c];
                NT[r] = AR[r * 3 + 0] * ti[0] + AR[r * 3 + 1] * ti[1] + AR[r * 3 + 2] * ti[2] + AT[r];
            }
            for (int q = 0; q < 9; ++q) AR[q] = NR[q];
            AT[0] = NT[0]; AT[1] = NT[1]; AT[2] = NT[2];
        }
    }
    // remove rest-pose transform from the translation column
    float Jj[3]; jloc(j, Jj);
    for (int r = 0; r < 3; ++r)
        AT[r] -= AR[r * 3 + 0] * Jj[0] + AR[r * 3 + 1] * Jj[1] + AR[r * 3 + 2] * Jj[2];

    float* g = G + (size_t)t * 12;
    for (int r = 0; r < 3; ++r) {
        g[r * 4 + 0] = AR[r * 3 + 0]; g[r * 4 + 1] = AR[r * 3 + 1];
        g[r * 4 + 2] = AR[r * 3 + 2]; g[r * 4 + 3] = AT[r];
    }
    // x = [beta(10) | lrotmin(207) | 0 pad(7)]
    float* x = xvec + (size_t)b * K1_;
    if (j == 0) {
        for (int s = 0; s < 10; ++s) x[s] = bet[s];
        for (int s = 217; s < K1_; ++s) x[s] = 0.0f;
    } else {
        for (int rc = 0; rc < 9; ++rc)
            x[10 + (j - 1) * 9 + rc] = Rj[rc] - ((rc % 4 == 0) ? 1.0f : 0.0f);
    }
}

// ---- 4) pack x vectors as GEMM1 B fragments ----
__global__ void k_pack_B(const float* __restrict__ xvec, __bf16* __restrict__ Bpack) {
    int t = blockIdx.x * blockDim.x + threadIdx.x;
    if (t >= BTILES_ * KT1_ * 512) return;
    int e = t & 15, l = (t >> 4) & 31, tile = t >> 9;
    int kt = tile % KT1_, bt = tile / KT1_;
    int half = l >> 4, n = l & 15;
    int gk = kt * 32 + frag_kB(e, half);
    Bpack[t] = f2bf(xvec[(size_t)(bt * 16 + n) * K1_ + gk]);
}

// ---- 5) pack skinning weights as GEMM2 A fragment (K=24 padded to 32) ----
__global__ void k_pack_W(const float* __restrict__ weights, __bf16* __restrict__ Wpack) {
    int t = blockIdx.x * blockDim.x + threadIdx.x;
    if (t >= VTILES_ * 512) return;
    int e = t & 15, l = (t >> 4) & 31, vt = t >> 9;
    int half = l >> 4, m = l & 15;
    int v = vt * 16 + m;
    int j = frag_kA(e, half);
    float w = (v < NV_ && j < NJ_) ? weights[(size_t)v * NJ_ + j] : 0.0f;
    Wpack[t] = f2bf(w);
}

// ---- 6) pack G columns (b,pq) as GEMM2 B fragments ----
__global__ void k_pack_G(const float* __restrict__ G, __bf16* __restrict__ Gpack) {
    int t = blockIdx.x * blockDim.x + threadIdx.x;
    if (t >= BTILES_ * 12 * 512) return;
    int e = t & 15, l = (t >> 4) & 31, tile = t >> 9;
    int nt = tile % 12, bt = tile / 12;
    int half = l >> 4, n = l & 15;
    int j = frag_kB(e, half);
    int c = nt * 16 + n;                  // column within the 192 (b_local,pq) cols
    int b = bt * 16 + c / 12, pq = c % 12;
    float v = (j < NJ_) ? G[((size_t)b * NJ_ + j) * 12 + pq] : 0.0f;
    Gpack[t] = f2bf(v);
}

// ---- 7) fused WMMA mega-kernel: one wave per 16-vertex x 16-batch tile ----
__global__ void __launch_bounds__(32)
k_mega(const float* __restrict__ vtemp, const float* __restrict__ transl,
       const __bf16* __restrict__ Apack_, const __bf16* __restrict__ Bpack_,
       const __bf16* __restrict__ Wpack_, const __bf16* __restrict__ Gpack_,
       float* __restrict__ out) {
    __shared__ float TS[16 * 192];   // T tile: 16 vertices x (16 b x 12 pq)
    __shared__ float VP[48 * 16];    // v_posed tile: 48 (v,d) rows x 16 batches
    const int vt = blockIdx.x;       // vertex tile  0..430
    const int bt = blockIdx.y;       // batch tile   0..63
    const int lid = threadIdx.x;
    const int half = lid >> 4, n = lid & 15;
    const int mrow = half * 8;

    const v16bf* Apack = (const v16bf*)Apack_;
    const v16bf* Bpack = (const v16bf*)Bpack_;
    const v16bf* Wpack = (const v16bf*)Wpack_;
    const v16bf* Gpack = (const v16bf*)Gpack_;

    // GEMM1: v_posed(48 x 16) = A(48 x 224) @ x(224 x 16), v_template added at store
    v8f acc0 = {}, acc1 = {}, acc2 = {};
    #pragma unroll
    for (int kt = 0; kt < KT1_; ++kt) {
        v16bf bf = Bpack[(bt * KT1_ + kt) * 32 + lid];
        const v16bf* ap = &Apack[((vt * 3 + 0) * KT1_ + kt) * 32 + lid];
        if (kt < KT1_ - 1)
            __builtin_prefetch((const void*)(ap + 32), 0, 1);   // next K-tile of A
        v16bf a0 = ap[0];
        v16bf a1 = ap[KT1_ * 32];
        v16bf a2 = ap[2 * KT1_ * 32];
        acc0 = __builtin_amdgcn_wmma_f32_16x16x32_bf16(false, a0, false, bf, (short)0, acc0, false, false);
        acc1 = __builtin_amdgcn_wmma_f32_16x16x32_bf16(false, a1, false, bf, (short)0, acc1, false, false);
        acc2 = __builtin_amdgcn_wmma_f32_16x16x32_bf16(false, a2, false, bf, (short)0, acc2, false, false);
    }
    #pragma unroll
    for (int r = 0; r < 8; ++r) {
        int m = r + mrow;
        int gm0 = vt * 48 + m, gm1 = gm0 + 16, gm2 = gm0 + 32;
        VP[(m)      * 16 + n] = acc0[r] + ((gm0 < M1_) ? vtemp[gm0] : 0.0f);
        VP[(m + 16) * 16 + n] = acc1[r] + ((gm1 < M1_) ? vtemp[gm1] : 0.0f);
        VP[(m + 32) * 16 + n] = acc2[r] + ((gm2 < M1_) ? vtemp[gm2] : 0.0f);
    }

    // GEMM2: T(16 x 192) = weights(16 x 32) @ G(32 x 192), K=24 zero-padded
    v16bf aw = Wpack[vt * 32 + lid];
    #pragma unroll
    for (int nt = 0; nt < 12; ++nt) {
        v16bf gb = Gpack[(bt * 12 + nt) * 32 + lid];
        v8f tacc = {};
        tacc = __builtin_amdgcn_wmma_f32_16x16x32_bf16(false, aw, false, gb, (short)0, tacc, false, false);
        #pragma unroll
        for (int r = 0; r < 8; ++r) {
            int m = r + mrow;
            TS[m * 192 + nt * 16 + n] = tacc[r];
        }
    }
    __syncthreads();

    // apply: out[b,v,p] = T[p,0:3] . v_posed + T[p,3] + transl[b,p]
    #pragma unroll
    for (int i = 0; i < 24; ++i) {
        int o = i * 32 + lid;            // 768 = 16v * 16b * 3p outputs per tile
        int p = o % 3, vb = o / 3;
        int vl = vb & 15, bl = vb >> 4;
        const float* Trow = &TS[vl * 192 + bl * 12 + p * 4];
        float r = Trow[3];
        r += Trow[0] * VP[(vl * 3 + 0) * 16 + bl];
        r += Trow[1] * VP[(vl * 3 + 1) * 16 + bl];
        r += Trow[2] * VP[(vl * 3 + 2) * 16 + bl];
        int gb = bt * 16 + bl, gv = vt * 16 + vl;
        if (gv < NV_)
            out[((size_t)gb * NV_ + gv) * 3 + p] = r + transl[gb * 3 + p];
    }
}

extern "C" void kernel_launch(void* const* d_in, const int* in_sizes, int n_in,
                              void* d_out, int out_size, void* d_ws, size_t ws_size,
                              hipStream_t stream) {
    (void)in_sizes; (void)n_in; (void)out_size; (void)ws_size;
    const float* pose      = (const float*)d_in[0];
    const float* beta      = (const float*)d_in[1];
    const float* transl    = (const float*)d_in[2];
    const float* vtemp     = (const float*)d_in[3];
    const float* shapedirs = (const float*)d_in[4];
    const float* posedirs  = (const float*)d_in[5];
    const float* Jreg      = (const float*)d_in[6];
    const float* weights   = (const float*)d_in[7];
    // d_in[8] = parents: compile-time constant tree, unused
    float* out = (float*)d_out;

    auto align256 = [](size_t x) { return (x + 255) & ~(size_t)255; };
    constexpr size_t SZ_A  = (size_t)MTILES_ * KT1_ * 512 * 2;   // 9.27 MB bf16
    constexpr size_t SZ_B  = (size_t)BTILES_ * KT1_ * 512 * 2;
    constexpr size_t SZ_W  = (size_t)VTILES_ * 512 * 2;
    constexpr size_t SZ_GF = (size_t)BTILES_ * 12 * 512 * 2;
    constexpr size_t SZ_G  = (size_t)NB_ * NJ_ * 12 * 4;
    constexpr size_t SZ_X  = (size_t)NB_ * K1_ * 4;

    char* ws = (char*)d_ws;
    size_t off = 0;
    __bf16* Apack = (__bf16*)(ws + off); off = align256(off + SZ_A);
    __bf16* Bpack = (__bf16*)(ws + off); off = align256(off + SZ_B);
    __bf16* Wpack = (__bf16*)(ws + off); off = align256(off + SZ_W);
    __bf16* Gpack = (__bf16*)(ws + off); off = align256(off + SZ_GF);
    float*  Gm    = (float*)(ws + off);  off = align256(off + SZ_G);
    float*  xvec  = (float*)(ws + off);  off = align256(off + SZ_X);
    float*  JTJS  = (float*)(ws + off);  // 792 floats

    k_prep_J<<<(NJ_ * 33 + 255) / 256, 256, 0, stream>>>(Jreg, vtemp, shapedirs, JTJS);
    k_pack_A<<<(MTILES_ * KT1_ * 512 + 255) / 256, 256, 0, stream>>>(shapedirs, posedirs, Apack);
    k_batch <<<(NB_ * NJ_ + 255) / 256, 256, 0, stream>>>(pose, beta, JTJS, Gm, xvec);
    k_pack_B<<<(BTILES_ * KT1_ * 512 + 255) / 256, 256, 0, stream>>>(xvec, Bpack);
    k_pack_W<<<(VTILES_ * 512 + 255) / 256, 256, 0, stream>>>(weights, Wpack);
    k_pack_G<<<(BTILES_ * 12 * 512 + 255) / 256, 256, 0, stream>>>(Gm, Gpack);
    dim3 grid(VTILES_, BTILES_);
    k_mega<<<grid, 32, 0, stream>>>(vtemp, transl, Apack, Bpack, Wpack, Gpack, out);
}
